// PCTransformer_37005438222870
// MI455X (gfx1250) — compile-verified, matrix operand order checked by reading, not run on hardware
//
#include <hip/hip_runtime.h>
#include <math.h>

// ---------------- problem constants (match reference) ----------------
constexpr int   kV = 16384, kD = 512, kF = 2048, kH = 8, kS = 512, kB = 2;
constexpr int   kDH = kD / kH;          // 64
constexpr int   kBS = kB * kS;          // 1024 rows
constexpr int   kT  = 3;
constexpr float kGamma = 0.1f;
constexpr float kScale = 0.125f;        // 1/sqrt(dh)

// ---------------- types ----------------
typedef __attribute__((ext_vector_type(16))) __bf16          v16bf;
typedef __attribute__((ext_vector_type(8)))  float           v8f;
typedef __attribute__((ext_vector_type(8)))  unsigned short  ushort8;  // 16B chunk

union FragAB { v16bf v; ushort8 h[2]; };

__device__ __forceinline__ unsigned short f2bf(float f) {
  union { float f; unsigned u; } c; c.f = f;
  unsigned u = c.u;
  u += 0x7fffu + ((u >> 16) & 1u);   // round-to-nearest-even
  return (unsigned short)(u >> 16);
}

// CDNA5 async global->LDS DMA (tracked by ASYNCcnt). Low 32 bits of a generic
// pointer to __shared__ are the DS address (ISA 10.2: LDS aperture truncation).
__device__ __forceinline__ void async_cp16(const void* g, unsigned lds) {
  asm volatile("global_load_async_to_lds_b128 %0, %1, off"
               :: "v"(lds), "v"(g) : "memory");
}
__device__ __forceinline__ void wait_async0() { asm volatile("s_wait_asynccnt 0" ::: "memory"); }
__device__ __forceinline__ void wait_async4() { asm volatile("s_wait_asynccnt 4" ::: "memory"); }

#define GF_BIAS 1
#define GF_ACC  2
#define GF_WF32 4
#define GF_WBF  8

// Strided-batched bf16 GEMM: C[z] = A[z] * B[z]^T' (+bias)(+C).
// A: bf16 [M][K] row-major (lda). B: bf16 stored [N][K] (ldb) -- i.e. logical
// KxN pre-transposed, so BOTH operands are contiguous along K (16B chunks).
// 128x128 tile, BK=32, 256 thr = 8 wave32 (4Mx2N); wave = 2x4 WMMA 16x16x32.
// Tiles staged via global_load_async_to_lds_b128, double-buffered (ASYNCcnt).
__global__ __launch_bounds__(256) void pc_gemm_bf16(
    const unsigned short* __restrict__ A, const unsigned short* __restrict__ B,
    const float* __restrict__ bias, float* __restrict__ C,
    unsigned short* __restrict__ Cbf,
    int M, int N, int K, int lda, int ldb, int ldc,
    long long aB1, long long aB2, long long bB1, long long bB2,
    long long cB1, long long cB2, int bdiv, int flags)
{
  __shared__ __align__(16) unsigned short As[2][128][40];  // [buf][row][32+8 pad]
  __shared__ __align__(16) unsigned short Bs[2][128][40];

  const int z = blockIdx.z;
  A += (long long)(z / bdiv) * aB1 + (long long)(z % bdiv) * aB2;
  B += (long long)(z / bdiv) * bB1 + (long long)(z % bdiv) * bB2;
  const long long coff = (long long)(z / bdiv) * cB1 + (long long)(z % bdiv) * cB2;
  if (flags & (GF_WF32 | GF_ACC)) C += coff;
  if (flags & GF_WBF)             Cbf += coff;

  const int tileN = blockIdx.x * 128, tileM = blockIdx.y * 128;
  const int t = threadIdx.x;
  const int lane = t & 31, wave = t >> 5;
  const int waveM = wave & 3, waveN = wave >> 2;
  const int l16 = lane & 15, hi = lane >> 4;

  v8f acc[2][4];
#pragma unroll
  for (int i = 0; i < 2; ++i)
#pragma unroll
    for (int j = 0; j < 4; ++j)
#pragma unroll
      for (int g = 0; g < 8; ++g) acc[i][j][g] = 0.f;

  // stage one 128x32 bf16 tile pair: 4x16B chunks per row, 2 rows per thread
  auto stage = [&](int buf, int k0) {
#pragma unroll
    for (int i = 0; i < 2; ++i) {
      int c = t + i * 256;
      int r = c >> 2, cc = (c & 3) * 8;
      async_cp16(A + (long long)(tileM + r) * lda + (k0 + cc),
                 (unsigned)(unsigned long long)&As[buf][r][cc]);
      async_cp16(B + (long long)(tileN + r) * ldb + (k0 + cc),
                 (unsigned)(unsigned long long)&Bs[buf][r][cc]);
    }
  };

  stage(0, 0);
  const int nk = K >> 5;
  for (int kt = 0; kt < nk; ++kt) {
    const int buf = kt & 1;
    if (kt + 1 < nk) { stage(buf ^ 1, (kt + 1) << 5); wait_async4(); }
    else             { wait_async0(); }
    __syncthreads();

    // Fragment = 2x ds_load_b128: row chunks {hi, 2+hi} give the CDNA5 16-bit
    // K-order (lanes 0-15: K 0-7,16-23; lanes 16-31: K 8-15,24-31).
    FragAB bfr[4];
#pragma unroll
    for (int tn = 0; tn < 4; ++tn) {
      const ushort8* r = (const ushort8*)&Bs[buf][waveN * 64 + tn * 16 + l16][0];
      bfr[tn].h[0] = r[hi];
      bfr[tn].h[1] = r[2 + hi];
    }
#pragma unroll
    for (int tm = 0; tm < 2; ++tm) {
      FragAB af;
      const ushort8* r = (const ushort8*)&As[buf][waveM * 32 + tm * 16 + l16][0];
      af.h[0] = r[hi];
      af.h[1] = r[2 + hi];
#pragma unroll
      for (int tn = 0; tn < 4; ++tn) {
        acc[tm][tn] = __builtin_amdgcn_wmma_f32_16x16x32_bf16(
            false, af.v, false, bfr[tn].v, (short)0, acc[tm][tn], false, false);
      }
    }
    __syncthreads();
  }

  // Epilogue: VGPR g -> M = g (lanes 0-15) / 8+g (lanes 16-31); N = lane&15.
#pragma unroll
  for (int tm = 0; tm < 2; ++tm) {
#pragma unroll
    for (int tn = 0; tn < 4; ++tn) {
      int col = tileN + waveN * 64 + tn * 16 + l16;
#pragma unroll
      for (int g = 0; g < 8; ++g) {
        int row = tileM + waveM * 32 + tm * 16 + hi * 8 + g;
        if (row < M && col < N) {
          float v = acc[tm][tn][g];
          if (flags & GF_BIAS) v += bias[col];
          long long idx = (long long)row * ldc + col;
          if (flags & GF_ACC)  v += C[idx];
          if (flags & GF_WF32) C[idx] = v;
          if (flags & GF_WBF)  Cbf[idx] = f2bf(v);
        }
      }
    }
  }
}

// ---------------- elementwise / reduction kernels ----------------
__device__ __forceinline__ float blk_sum(float v, float* sm) {
  int t = threadIdx.x;
  sm[t] = v; __syncthreads();
  for (int s = 128; s > 0; s >>= 1) { if (t < s) sm[t] += sm[t + s]; __syncthreads(); }
  float r = sm[0]; __syncthreads(); return r;
}
__device__ __forceinline__ float blk_max(float v, float* sm) {
  int t = threadIdx.x;
  sm[t] = v; __syncthreads();
  for (int s = 128; s > 0; s >>= 1) { if (t < s) sm[t] = fmaxf(sm[t], sm[t + s]); __syncthreads(); }
  float r = sm[0]; __syncthreads(); return r;
}

// f32 weight (R,C) -> bf16 original layout + bf16 transposed (C,R)
__global__ void pc_wcvt(const float* __restrict__ w, unsigned short* __restrict__ o,
                        unsigned short* __restrict__ tr, long long R, long long C) {
  long long i = (long long)blockIdx.x * 256 + threadIdx.x;
  if (i < R * C) {
    long long r = i / C, c = i - r * C;
    unsigned short h = f2bf(w[i]);
    o[i] = h;
    tr[c * R + r] = h;
  }
}

// bf16 (B,S,H*dh) -> bf16 (B,H,dh,S)
__global__ void pc_tr_heads(const unsigned short* __restrict__ in,
                            unsigned short* __restrict__ out) {
  long long i = (long long)blockIdx.x * 256 + threadIdx.x;
  if (i < (long long)kBS * kD) {
    long long row = i / kD; int col = (int)(i - row * kD);
    int b = (int)(row / kS), s = (int)(row % kS);
    int h = col / kDH, d = col - h * kDH;
    out[((long long)(b * kH + h) * kDH + d) * kS + s] = in[i];
  }
}

__global__ void pc_embed(const float* __restrict__ we, const float* __restrict__ pe,
                         const int* __restrict__ ids, float* __restrict__ o0,
                         float* __restrict__ o1) {
  long long row = blockIdx.x;
  int id = ids[row]; id = id < 0 ? 0 : (id > kV - 1 ? kV - 1 : id);
  int sr = (int)(row % kS);
  for (int d = threadIdx.x; d < kD; d += 256) {
    float v = we[(long long)id * kD + d] + pe[(long long)sr * kD + d];
    o0[row * kD + d] = v; o1[row * kD + d] = v;
  }
}

__global__ void pc_ln_fwd(const float* __restrict__ x, const float* __restrict__ sc,
                          const float* __restrict__ bi, unsigned short* __restrict__ y) {
  __shared__ float sm[256];
  long long row = blockIdx.x;
  const float* xr = x + row * kD;
  float s = 0.f;
  for (int d = threadIdx.x; d < kD; d += 256) s += xr[d];
  float mean = blk_sum(s, sm) / kD;
  float v = 0.f;
  for (int d = threadIdx.x; d < kD; d += 256) { float t = xr[d] - mean; v += t * t; }
  float rstd = rsqrtf(blk_sum(v, sm) / kD + 1e-5f);
  for (int d = threadIdx.x; d < kD; d += 256)
    y[row * kD + d] = f2bf((xr[d] - mean) * rstd * sc[d] + bi[d]);
}

__global__ void pc_ln_vjp(const float* __restrict__ x, const float* __restrict__ dy,
                          const float* __restrict__ sc, float* __restrict__ dx) {
  __shared__ float sm[256];
  long long row = blockIdx.x;
  const float* xr = x + row * kD;
  const float* dr = dy + row * kD;
  float s = 0.f;
  for (int d = threadIdx.x; d < kD; d += 256) s += xr[d];
  float mean = blk_sum(s, sm) / kD;
  float v = 0.f;
  for (int d = threadIdx.x; d < kD; d += 256) { float t = xr[d] - mean; v += t * t; }
  float rstd = rsqrtf(blk_sum(v, sm) / kD + 1e-5f);
  float s1 = 0.f, s2 = 0.f;
  for (int d = threadIdx.x; d < kD; d += 256) {
    float a = dr[d] * sc[d];
    float xh = (xr[d] - mean) * rstd;
    s1 += a; s2 += a * xh;
  }
  s1 = blk_sum(s1, sm) / kD;
  s2 = blk_sum(s2, sm) / kD;
  for (int d = threadIdx.x; d < kD; d += 256) {
    float a = dr[d] * sc[d];
    float xh = (xr[d] - mean) * rstd;
    dx[row * kD + d] = rstd * (a - s1 - xh * s2);
  }
}

// causal softmax (scale folded); keeps f32 P, writes bf16 P and bf16 P^T
__global__ void pc_softmax_fwd(float* __restrict__ P, unsigned short* __restrict__ Pbf,
                               unsigned short* __restrict__ Pt) {
  __shared__ float sm[256];
  long long row = blockIdx.x;
  long long zz = row / kS; int q = (int)(row - zz * kS);
  float* p = P + row * kS;
  float mx = -3.0e38f;
  for (int j = threadIdx.x; j < kS; j += 256)
    if (j <= q) mx = fmaxf(mx, p[j] * kScale);
  mx = blk_max(mx, sm);
  float sum = 0.f;
  for (int j = threadIdx.x; j < kS; j += 256)
    if (j <= q) sum += __expf(p[j] * kScale - mx);
  sum = blk_sum(sum, sm);
  float inv = 1.f / sum;
  for (int j = threadIdx.x; j < kS; j += 256) {
    float v = (j <= q) ? __expf(p[j] * kScale - mx) * inv : 0.f;
    p[j] = v;
    unsigned short h = f2bf(v);
    Pbf[row * kS + j] = h;
    Pt[zz * (long long)kS * kS + (long long)j * kS + q] = h;
  }
}

// softmax VJP (scale folded): dZ = P*(dP - sum(P*dP))*scale -> bf16 dZ, dZ^T
__global__ void pc_softmax_bwd(const float* __restrict__ P, const float* __restrict__ dP,
                               unsigned short* __restrict__ Zbf,
                               unsigned short* __restrict__ Zt) {
  __shared__ float sm[256];
  long long row = blockIdx.x;
  long long zz = row / kS; int q = (int)(row - zz * kS);
  const float* p = P + row * kS;
  const float* dp = dP + row * kS;
  float dot = 0.f;
  for (int j = threadIdx.x; j < kS; j += 256) dot += p[j] * dp[j];
  dot = blk_sum(dot, sm);
  for (int j = threadIdx.x; j < kS; j += 256) {
    unsigned short h = f2bf(p[j] * (dp[j] - dot) * kScale);
    Zbf[row * kS + j] = h;
    Zt[zz * (long long)kS * kS + (long long)j * kS + q] = h;
  }
}

__global__ void pc_gelu_fwd(const float* __restrict__ x, float* __restrict__ y,
                            unsigned short* __restrict__ ybf, long long n) {
  long long i = (long long)blockIdx.x * 256 + threadIdx.x;
  if (i < n) {
    float v = x[i];
    float u = 0.7978845608028654f * (v + 0.044715f * v * v * v);
    float r = 0.5f * v * (1.f + tanhf(u));
    y[i] = r;
    if (ybf) ybf[i] = f2bf(r);
  }
}

__global__ void pc_gelu_bwd(const float* __restrict__ x, const float* __restrict__ e,
                            unsigned short* __restrict__ o, long long n) {
  long long i = (long long)blockIdx.x * 256 + threadIdx.x;
  if (i < n) {
    float v = x[i];
    float u = 0.7978845608028654f * (v + 0.044715f * v * v * v);
    float th = tanhf(u);
    float du = 0.7978845608028654f * (1.f + 3.f * 0.044715f * v * v);
    o[i] = f2bf(e[i] * (0.5f * (1.f + th) + 0.5f * v * (1.f - th * th) * du));
  }
}

__global__ void pc_eps(float* __restrict__ e, unsigned short* __restrict__ ebf,
                       const float* __restrict__ x, const float* __restrict__ mu,
                       long long n) {
  long long i = (long long)blockIdx.x * 256 + threadIdx.x;
  if (i < n) { float v = x[i] - mu[i]; e[i] = v; ebf[i] = f2bf(v); }
}

__global__ void pc_update(float* __restrict__ x, unsigned short* __restrict__ xbf,
                          const float* __restrict__ td, const float* __restrict__ e,
                          long long n) {
  long long i = (long long)blockIdx.x * 256 + threadIdx.x;
  if (i < n) {
    float v = x[i] + kGamma * (td[i] - e[i]);
    x[i] = v;
    if (xbf) xbf[i] = f2bf(v);
  }
}

__global__ void pc_epstop(float* __restrict__ L, unsigned short* __restrict__ Lbf,
                          const int* __restrict__ tg, long long n) {
  long long i = (long long)blockIdx.x * 256 + threadIdx.x;
  if (i < n) {
    long long row = i / kV; int col = (int)(i - row * kV);
    int t = tg[row]; t = t < 0 ? 0 : (t > kV - 1 ? kV - 1 : t);
    float v = L[i];
    if (col == t) v -= 1.f;
    L[i] = v;
    Lbf[i] = f2bf(v);
  }
}

// ---------------- host orchestration ----------------
extern "C" void kernel_launch(void* const* d_in, const int* in_sizes, int n_in,
                              void* d_out, int out_size, void* d_ws, size_t ws_size,
                              hipStream_t stream) {
  (void)in_sizes; (void)n_in; (void)out_size; (void)ws_size;
  const float* word_emb = (const float*)d_in[0];
  const float* pos_emb  = (const float*)d_in[1];
  const float* Wq = (const float*)d_in[2];
  const float* Wk = (const float*)d_in[3];
  const float* Wv = (const float*)d_in[4];
  const float* bq = (const float*)d_in[5];
  const float* bk = (const float*)d_in[6];
  const float* bv = (const float*)d_in[7];
  const float* Wo = (const float*)d_in[8];
  const float* bo = (const float*)d_in[9];
  const float* ln1_s = (const float*)d_in[10];
  const float* ln1_b = (const float*)d_in[11];
  const float* ln2_s = (const float*)d_in[12];
  const float* ln2_b = (const float*)d_in[13];
  const float* W1 = (const float*)d_in[14];
  const float* b1 = (const float*)d_in[15];
  const float* W2 = (const float*)d_in[16];
  const float* b2 = (const float*)d_in[17];
  const float* Wout = (const float*)d_in[18];
  const float* bout = (const float*)d_in[19];
  const int* ids = (const int*)d_in[20];
  const int* tgt = (const int*)d_in[21];
  float* out = (float*)d_out;
  typedef unsigned short ush;

  // ---- workspace layout (256B aligned) ----
  char* wsb = (char*)d_ws; size_t off = 0;
  auto ALF = [&](size_t n) { float* p = (float*)(wsb + off); off += ((n * 4 + 255) & ~(size_t)255); return p; };
  auto ALH = [&](size_t n) { ush* p = (ush*)(wsb + off); off += ((n * 2 + 255) & ~(size_t)255); return p; };
  const size_t nD = (size_t)kBS * kD, nF = (size_t)kBS * kF;
  const size_t nP = (size_t)kB * kH * kS * kS, nL = (size_t)kBS * kV;
  const long long DD2 = (long long)kD * kD, DF = (long long)kD * kF;
  const long long FD = (long long)kF * kD, DV = (long long)kD * kV;

  float* MU = ALF(nD);
  float* xs[17];
  for (int l = 0; l <= 16; ++l) xs[l] = ALF(((l & 3) == 3) ? nF : nD);
  float* EPSp = ALF(nF); float* EPSn = ALF(nF);
  float* TD = ALF(nF); float* PRED = ALF(nF);
  float* Pp = ALF(nP); float* dPp = ALF(nP);
  float* PRE = ALF(nF); float* LOG = ALF(nL);

  // bf16 weight mirrors: *_o original layout (for VJP, stored [N][K]),
  //                      *_t transposed (for forward, stored [N][K]).
  ush* wq_o = ALH(4 * DD2); ush* wq_t = ALH(4 * DD2);
  ush* wk_o = ALH(4 * DD2); ush* wk_t = ALH(4 * DD2);
  ush* wv_o = ALH(4 * DD2); ush* wv_t = ALH(4 * DD2);
  ush* wo_o = ALH(4 * DD2); ush* wo_t = ALH(4 * DD2);
  ush* w1_o = ALH(4 * DF);  ush* w1_t = ALH(4 * DF);
  ush* w2_o = ALH(4 * FD);  ush* w2_t = ALH(4 * FD);
  ush* wout_o = ALH(DV);    ush* wout_t = ALH(DV);
  // bf16 activations
  ush* H1bf = ALH(nD);
  ush* Qbf = ALH(nD); ush* Kbf = ALH(nD); ush* Vbf = ALH(nD);
  ush* Qt  = ALH(nD); ush* Kt  = ALH(nD); ush* Vt  = ALH(nD);
  ush* EBF = ALH(nF); ush* ET = ALH(nD);
  ush* Pbf = ALH(nP); ush* Pt = ALH(nP);
  ush* Zbf = ALH(nP); ush* Zt = ALH(nP);
  ush* DPREbf = ALH(nF);
  ush* DQbf = ALH(nD); ush* DKbf = ALH(nD); ush* DVbf = ALH(nD);
  ush* LOGbf = ALH(nL);
  ush* xbf[17];
  for (int l = 0; l <= 16; ++l)
    xbf[l] = ((l & 1) == 1 || l == 16) ? ALH(((l & 3) == 3) ? nF : nD) : nullptr;
  (void)ALH(65536);  // tail pad: N=64 GEMMs read <=64 B-rows past a slice

  const long long SD = (long long)kS * kD, SSe = (long long)kS * kS;
  const long long HSS = (long long)kH * SSe;
  const long long dhS = (long long)kDH * kS, HdhS = (long long)kH * dhS;

  auto g1 = [](long long n) { return dim3((unsigned)((n + 255) / 256)); };

  auto gemm = [&](const ush* A, int lda, const ush* B, int ldb,
                  float* C, ush* Cbf, int ldc, int M, int N, int K,
                  const float* bias, int flags,
                  int batch = 1, int bdiv = 1,
                  long long a1 = 0, long long a2 = 0,
                  long long b1_ = 0, long long b2_ = 0,
                  long long c1 = 0, long long c2 = 0) {
    dim3 grid((N + 127) / 128, (M + 127) / 128, batch);
    pc_gemm_bf16<<<grid, 256, 0, stream>>>(A, B, bias, C, Cbf, M, N, K, lda, ldb, ldc,
                                           a1, a2, b1_, b2_, c1, c2, bdiv, flags);
  };

  // ---- weight bf16 mirrors (deterministic; every call) ----
  for (int b = 0; b < 4; ++b) {
    pc_wcvt<<<g1(DD2), 256, 0, stream>>>(Wq + b * DD2, wq_o + b * DD2, wq_t + b * DD2, kD, kD);
    pc_wcvt<<<g1(DD2), 256, 0, stream>>>(Wk + b * DD2, wk_o + b * DD2, wk_t + b * DD2, kD, kD);
    pc_wcvt<<<g1(DD2), 256, 0, stream>>>(Wv + b * DD2, wv_o + b * DD2, wv_t + b * DD2, kD, kD);
    pc_wcvt<<<g1(DD2), 256, 0, stream>>>(Wo + b * DD2, wo_o + b * DD2, wo_t + b * DD2, kD, kD);
    pc_wcvt<<<g1(DF), 256, 0, stream>>>(W1 + b * DF, w1_o + b * DF, w1_t + b * DF, kD, kF);
    pc_wcvt<<<g1(FD), 256, 0, stream>>>(W2 + b * FD, w2_o + b * FD, w2_t + b * FD, kF, kD);
  }
  pc_wcvt<<<g1(DV), 256, 0, stream>>>(Wout, wout_o, wout_t, kD, kV);

  // forward of layer l: xin (f32, xinbf for linear-input layers) -> outf/outbf
  auto fwd = [&](int l, const float* xin, const ush* xinbf, float* outf, ush* outbf) {
    int b = l >> 2, type = l & 3;
    int wb = (outbf ? GF_WBF : 0);
    switch (type) {
      case 0: {
        pc_ln_fwd<<<kBS, 256, 0, stream>>>(xin, ln1_s + b * kD, ln1_b + b * kD, H1bf);
        gemm(H1bf, kD, wq_t + b * DD2, kD, nullptr, Qbf, kD, kBS, kD, kD, bq + b * kD, GF_BIAS | GF_WBF);
        gemm(H1bf, kD, wk_t + b * DD2, kD, nullptr, Kbf, kD, kBS, kD, kD, bk + b * kD, GF_BIAS | GF_WBF);
        gemm(H1bf, kD, wv_t + b * DD2, kD, nullptr, Vbf, kD, kBS, kD, kD, bv + b * kD, GF_BIAS | GF_WBF);
        pc_tr_heads<<<g1(nD), 256, 0, stream>>>(Qbf, Qt);
        pc_tr_heads<<<g1(nD), 256, 0, stream>>>(Kbf, Kt);
        pc_tr_heads<<<g1(nD), 256, 0, stream>>>(Vbf, Vt);
        // scores: A=Q [S][dh], B=K stored [key][dh]  -> P (B,H,S,S) f32
        gemm(Qbf, kD, Kbf, kD, Pp, nullptr, kS, kS, kS, kDH, nullptr, GF_WF32,
             kB * kH, kH, SD, kDH, SD, kDH, HSS, SSe);
        pc_softmax_fwd<<<kB * kH * kS, 256, 0, stream>>>(Pp, Pbf, Pt);
        // ctx: A=P [q][key], B=V^T stored [d][key]
        gemm(Pbf, kS, Vt, kS, outf, outbf, kD, kS, kDH, kS, nullptr, GF_WF32 | wb,
             kB * kH, kH, HSS, SSe, HdhS, dhS, SD, kDH);
        break;
      }
      case 1:
        gemm(xinbf, kD, wo_t + b * DD2, kD, outf, outbf, kD, kBS, kD, kD,
             bo + b * kD, GF_BIAS | GF_WF32 | wb);
        break;
      case 2: {
        pc_ln_fwd<<<kBS, 256, 0, stream>>>(xin, ln2_s + b * kD, ln2_b + b * kD, H1bf);
        gemm(H1bf, kD, w1_t + b * DF, kD, PRE, nullptr, kF, kBS, kF, kD,
             b1 + b * kF, GF_BIAS | GF_WF32);
        pc_gelu_fwd<<<g1(nF), 256, 0, stream>>>(PRE, outf, outbf, (long long)nF);
        break;
      }
      case 3:
        gemm(xinbf, kF, w2_t + b * FD, kF, outf, outbf, kD, kBS, kD, kF,
             b2 + b * kD, GF_BIAS | GF_WF32 | wb);
        break;
    }
  };

  // VJP of layer l at xin with cotangent (epsN f32, ebf bf16) -> TD (f32)
  auto vjp = [&](int l, const float* xin, const float* epsN, const ush* ebf) {
    int b = l >> 2, type = l & 3;
    switch (type) {
      case 0: {
        // dP = dctx * V^T : B=V stored [key][dh]
        gemm(ebf, kD, Vbf, kD, dPp, nullptr, kS, kS, kS, kDH, nullptr, GF_WF32,
             kB * kH, kH, SD, kDH, SD, kDH, HSS, SSe);
        pc_softmax_bwd<<<kB * kH * kS, 256, 0, stream>>>(Pp, dPp, Zbf, Zt);
        pc_tr_heads<<<g1(nD), 256, 0, stream>>>(ebf, ET);
        // dV = P^T * dctx ; dQ = dZ * K ; dK = dZ^T * Q  (all NN-A / TB-B)
        gemm(Pt, kS, ET, kS, nullptr, DVbf, kD, kS, kDH, kS, nullptr, GF_WBF,
             kB * kH, kH, HSS, SSe, HdhS, dhS, SD, kDH);
        gemm(Zbf, kS, Kt, kS, nullptr, DQbf, kD, kS, kDH, kS, nullptr, GF_WBF,
             kB * kH, kH, HSS, SSe, HdhS, dhS, SD, kDH);
        gemm(Zt, kS, Qt, kS, nullptr, DKbf, kD, kS, kDH, kS, nullptr, GF_WBF,
             kB * kH, kH, HSS, SSe, HdhS, dhS, SD, kDH);
        // dh = dQ Wq^T + dK Wk^T + dV Wv^T  (B = original-layout mirrors)
        gemm(DQbf, kD, wq_o + b * DD2, kD, TD, nullptr, kD, kBS, kD, kD, nullptr, GF_WF32);
        gemm(DKbf, kD, wk_o + b * DD2, kD, TD, nullptr, kD, kBS, kD, kD, nullptr, GF_WF32 | GF_ACC);
        gemm(DVbf, kD, wv_o + b * DD2, kD, TD, nullptr, kD, kBS, kD, kD, nullptr, GF_WF32 | GF_ACC);
        pc_ln_vjp<<<kBS, 256, 0, stream>>>(xin, TD, ln1_s + b * kD, TD);
        break;
      }
      case 1:
        gemm(ebf, kD, wo_o + b * DD2, kD, TD, nullptr, kD, kBS, kD, kD, nullptr, GF_WF32);
        break;
      case 2: {
        pc_gelu_bwd<<<g1(nF), 256, 0, stream>>>(PRE, epsN, DPREbf, (long long)nF);
        gemm(DPREbf, kF, w1_o + b * DF, kF, TD, nullptr, kD, kBS, kD, kF, nullptr, GF_WF32);
        pc_ln_vjp<<<kBS, 256, 0, stream>>>(xin, TD, ln2_s + b * kD, TD);
        break;
      }
      case 3:
        gemm(ebf, kD, w2_o + b * FD, kD, TD, nullptr, kF, kBS, kF, kD, nullptr, GF_WF32);
        break;
    }
  };

  auto dimOf = [&](int l) -> long long { return ((l & 3) == 3) ? (long long)nF : (long long)nD; };

  // ---- init: embeddings + forward sweep ----
  pc_embed<<<kBS, 256, 0, stream>>>(word_emb, pos_emb, ids, MU, xs[0]);
  for (int l = 0; l < 16; ++l) fwd(l, xs[l], xbf[l], xs[l + 1], xbf[l + 1]);

  // ---- T PC iterations (streamed, in-place latent updates) ----
  float* ep = EPSp; float* en = EPSn;
  for (int it = 0; it < kT; ++it) {
    pc_eps<<<g1(nD), 256, 0, stream>>>(ep, EBF, xs[0], MU, (long long)nD);
    for (int l = 0; l < 16; ++l) {
      fwd(l, xs[l], xbf[l], PRED, nullptr);
      long long nNext = dimOf(l + 1);
      pc_eps<<<g1(nNext), 256, 0, stream>>>(en, EBF, xs[l + 1], PRED, nNext);
      vjp(l, xs[l], en, EBF);
      long long nCur = dimOf(l);
      pc_update<<<g1(nCur), 256, 0, stream>>>(xs[l], xbf[l], TD, ep, nCur);
      float* tswap = ep; ep = en; en = tswap;
    }
    // top: logits, eps_top = logits - onehot, td = eps_top @ Wout^T
    gemm(xbf[16], kD, wout_t, kD, LOG, nullptr, kV, kBS, kV, kD, bout, GF_BIAS | GF_WF32);
    pc_epstop<<<g1((long long)nL), 256, 0, stream>>>(LOG, LOGbf, tgt, (long long)nL);
    gemm(LOGbf, kV, wout_o, kV, TD, nullptr, kD, kBS, kD, kV, nullptr, GF_WF32);
    pc_update<<<g1(nD), 256, 0, stream>>>(xs[16], xbf[16], TD, ep, (long long)nD);
  }

  // ---- final output: logits of updated top latent ----
  gemm(xbf[16], kD, wout_t, kD, out, nullptr, kV, kBS, kV, kD, bout, GF_BIAS | GF_WF32);
}
